// VQ_30253749633315
// MI455X (gfx1250) — matrix operand order repaired and checked by reading, not compiled, not run on hardware
//
#include <hip/hip_runtime.h>

// ---------------------------------------------------------------------------
// VQ-VAE codebook quantization for MI455X (gfx1250, wave32, WMMA).
//   dist = ||e||^2 - 2 * x . e      (argmin over 1024 codes, f16 WMMA matmul)
//   loss = 1.25 * mean((quant - x)^2)
//   q_out = gathered codebook rows (NCHW), index_e = one-hot, encoded_e = idx
// ---------------------------------------------------------------------------

typedef __attribute__((ext_vector_type(16))) _Float16 v16h;
typedef __attribute__((ext_vector_type(8)))  _Float16 v8h;
typedef __attribute__((ext_vector_type(2)))  _Float16 v2h;
typedef __attribute__((ext_vector_type(8)))  float    v8f;
typedef __attribute__((ext_vector_type(4)))  float    v4f;
typedef __attribute__((ext_vector_type(2)))  float    v2f;

#define LEN_E   1024      // codebook size
#define SIZE_E  256       // embedding / channel dim
#define HW      4096      // 64*64 pixels per image
#define NPIX    65536     // 16 * HW
#define ROWS    128       // pixels handled per block
#define CHUNK   128       // codes per LDS chunk (8 chunks total)
#define LSTR    264       // LDS row stride in halves: 132 dwords -> bank step 4
                          // -> conflict-free 16-lane b128 reads; 528B = 33*16B

#define Q_ELEMS   16777216ull   // 16*256*64*64
#define IDX_ELEMS 67108864ull   // 65536*1024

union AF { struct { v8h lo, hi; } p; v16h v; };

// ---------------------------------------------------------------------------
// Kernel 0: precompute ||e||^2 per code into d_ws; zero the loss accumulator.
// ---------------------------------------------------------------------------
__global__ void vq_prep(const float* __restrict__ emb,
                        float* __restrict__ enorm,
                        float* __restrict__ loss_out) {
    int k = blockIdx.x * blockDim.x + threadIdx.x;   // 0..1023
    if (k == 0) loss_out[0] = 0.0f;
    if (k < LEN_E) {
        const v4f* er = (const v4f*)(emb + (size_t)k * SIZE_E);
        float s = 0.0f;
#pragma unroll 8
        for (int i = 0; i < SIZE_E / 4; ++i) {
            v4f e = er[i];
            s += e[0]*e[0] + e[1]*e[1] + e[2]*e[2] + e[3]*e[3];
        }
        enorm[k] = s;
    }
}

// ---------------------------------------------------------------------------
// Kernel 1: main fused kernel. 512 blocks x 256 threads (8 waves).
// Block bb owns pixels [bb*128, bb*128+128) (one image slice: img = bb>>5).
// ---------------------------------------------------------------------------
__global__ void __launch_bounds__(256)
vq_main(const float* __restrict__ x,      // [16][256][4096] NCHW
        const float* __restrict__ emb,    // [1024][256]
        const float* __restrict__ enorm,  // [1024]
        float* __restrict__ out) {        // loss | q_out | index_e | encoded_e

    __shared__ __align__(16) _Float16 xt[ROWS * LSTR];   // x tile, transposed, f16
    __shared__ __align__(16) _Float16 et[CHUNK * LSTR];  // emb chunk, f16
    __shared__ float redv[8 * 8 * 32];                   // argmin cross-lane scratch
    __shared__ int   redi[8 * 8 * 32];
    __shared__ int   enc[ROWS];
    __shared__ float lsum;

    const int tid  = threadIdx.x;
    const int bb   = blockIdx.x;         // 0..511
    const int img  = bb >> 5;            // 32 blocks per image
    const int p0   = (bb & 31) * ROWS;   // pixel offset inside the image
    const float* xbase = x + (size_t)img * SIZE_E * HW;

    // ---- stage x tile: [p][c] f16, transposed from NCHW (coalesced over p) ----
    for (int idx = tid; idx < ROWS * SIZE_E; idx += 256) {
        int c = idx >> 7;            // 0..255
        int p = idx & (ROWS - 1);    // 0..127
        xt[p * LSTR + c] = (_Float16)xbase[(size_t)c * HW + p0 + p];
    }

    const int wave  = tid >> 5;
    const int lane  = tid & 31;
    const int r0    = wave * 16;         // this wave's 16-row slab
    const int mrow  = lane & 15;
    const int khalf = lane >> 4;         // ISA A/B lane-half K selector

    float bestv[8];
    int   besti[8];
#pragma unroll
    for (int i = 0; i < 8; ++i) { bestv[i] = 3.0e38f; besti[i] = 0; }

    // ---- loop over 8 chunks of 128 codes ----
    for (int ch = 0; ch < 8; ++ch) {
        __syncthreads();   // xt staged (first iter) / previous chunk consumed
        {   // stage emb chunk: et[n][c] f16, coalesced row reads (float2)
            const float* erow = emb + (size_t)(ch * CHUNK) * SIZE_E;
            for (int idx = tid; idx < CHUNK * (SIZE_E / 2); idx += 256) {
                int n  = idx >> 7;             // 0..127
                int c2 = (idx & 127) * 2;      // 0,2,...,254
                v2f e = *(const v2f*)(erow + (size_t)n * SIZE_E + c2);
                v2h h; h[0] = (_Float16)e[0]; h[1] = (_Float16)e[1];
                *(v2h*)(et + n * LSTR + c2) = h;
            }
        }
        __syncthreads();

        // A fragments for this wave's 16 rows (ISA 7.12.2 16-bit A 16x32 layout):
        // lane L: row = lane&15; lanes 0-15 hold K {kk*32+0..7, +16..23},
        //         lanes 16-31 hold K {kk*32+8..15, +24..31}.
        AF a[8];
        const _Float16* arow = xt + (r0 + mrow) * LSTR;
#pragma unroll
        for (int kk = 0; kk < 8; ++kk) {
            const _Float16* ab = arow + kk * 32 + khalf * 8;
            a[kk].p.lo = *(const v8h*)(ab);
            a[kk].p.hi = *(const v8h*)(ab + 16);
        }

        const int n0 = ch * CHUNK;
#pragma unroll
        for (int cs = 0; cs < 8; ++cs) {
            // B fragment (32x16): lane L -> col N = lane&15; lanes 0-15 hold K {0..7, 16..23},
            // lanes 16-31 hold K {8..15, 24..31} (matches A) -> two 16B LDS reads.
            const int col = cs * 16 + mrow;
            const _Float16* brow = et + col * LSTR + khalf * 8;
            v8f acc = {0.f,0.f,0.f,0.f,0.f,0.f,0.f,0.f};
#pragma unroll
            for (int kk = 0; kk < 8; ++kk) {
                AF b;
                b.p.lo = *(const v8h*)(brow + kk * 32);
                b.p.hi = *(const v8h*)(brow + kk * 32 + 16);
                acc = __builtin_amdgcn_wmma_f32_16x16x32_f16(
                          false, a[kk].v, false, b.v,
                          (short)0, acc, false, false);
            }
            // C/D layout: vgpr i -> row r0 + i + (lane>>4)*8, col = cs*16+(lane&15)
            const int code = n0 + col;
            const float en = enorm[code];
#pragma unroll
            for (int i = 0; i < 8; ++i) {
                float d = en - 2.0f * acc[i];
                if (d < bestv[i]) { bestv[i] = d; besti[i] = code; }
            }
        }
    }

    // ---- cross-lane argmin: row r0+i lives in lanes 0-15 (i<8) / 16-31 (i+8) ----
#pragma unroll
    for (int i = 0; i < 8; ++i) {
        redv[(wave * 8 + i) * 32 + lane] = bestv[i];
        redi[(wave * 8 + i) * 32 + lane] = besti[i];
    }
    if (tid == 0) lsum = 0.0f;
    __syncthreads();
    if (lane < 16) {
        int r = lane;                 // local row 0..15 in this wave's slab
        int i = r & 7, h = r >> 3;
        float bv = 3.0e38f; int bi = 0x7fffffff;
        for (int j = 0; j < 16; ++j) {
            float v  = redv[(wave * 8 + i) * 32 + h * 16 + j];
            int   ix = redi[(wave * 8 + i) * 32 + h * 16 + j];
            if (v < bv || (v == bv && ix < bi)) { bv = v; bi = ix; }
        }
        enc[r0 + r] = bi;
    }
    __syncthreads();

    // ---- outputs ----
    float* d_q   = out + 1;
    float* d_idx = out + 1 + Q_ELEMS;
    float* d_enc = out + 1 + Q_ELEMS + IDX_ELEMS;
    const size_t grow0 = (size_t)bb * ROWS;

    if (tid < ROWS)
        __builtin_nontemporal_store((float)enc[tid], d_enc + grow0 + tid);

    // one-hot index_e rows (scalar NT stores: base offset is 4B-aligned only)
    for (int idx = tid; idx < ROWS * LEN_E; idx += 256) {
        int p   = idx >> 10;
        int col = idx & (LEN_E - 1);
        float v = (col == enc[p]) ? 1.0f : 0.0f;
        __builtin_nontemporal_store(v, d_idx + grow0 * LEN_E + idx);
    }

    // quantized output (NCHW) + loss partial; x re-read hits L2 (x fits in 192MB)
    float lacc = 0.0f;
    for (int idx = tid; idx < ROWS * SIZE_E; idx += 256) {
        int c = idx >> 7;
        int p = idx & (ROWS - 1);
        int e = enc[p];
        float qv = emb[(size_t)e * SIZE_E + c];
        float xv = xbase[(size_t)c * HW + p0 + p];
        float dd = qv - xv;
        lacc += dd * dd;
        __builtin_nontemporal_store(qv, d_q + (size_t)img * SIZE_E * HW
                                           + (size_t)c * HW + p0 + p);
    }
#pragma unroll
    for (int off = 16; off > 0; off >>= 1)
        lacc += __shfl_xor(lacc, off, 32);
    if (lane == 0) atomicAdd(&lsum, lacc);
    __syncthreads();
    if (tid == 0)
        atomicAdd(out, lsum * (1.25f / 16777216.0f));  // (1+beta)/numel
}

// ---------------------------------------------------------------------------
extern "C" void kernel_launch(void* const* d_in, const int* in_sizes, int n_in,
                              void* d_out, int out_size, void* d_ws, size_t ws_size,
                              hipStream_t stream) {
    const float* x   = (const float*)d_in[0];   // [16,256,64,64]
    const float* emb = (const float*)d_in[1];   // [1024,256]
    float* out   = (float*)d_out;               // 1 + 16777216 + 67108864 + 65536
    float* enorm = (float*)d_ws;                // 4 KB scratch

    vq_prep<<<4, 256, 0, stream>>>(emb, enorm, out);
    vq_main<<<NPIX / ROWS, 256, 0, stream>>>(x, emb, enorm, out);
}
